// MPSLayer_12326556140216
// MI455X (gfx1250) — matrix operand order repaired and checked by reading, not compiled, not run on hardware
//
#include <hip/hip_runtime.h>

typedef _Float16 hf;
typedef __attribute__((ext_vector_type(16))) _Float16 v16h;
typedef __attribute__((ext_vector_type(8)))  _Float16 v8h;
typedef __attribute__((ext_vector_type(8)))  float    v8f;
typedef __attribute__((ext_vector_type(4)))  float    v4f;
typedef __attribute__((ext_vector_type(4)))  unsigned int v4u;
typedef __attribute__((ext_vector_type(8)))  int      v8i;
typedef __attribute__((ext_vector_type(4)))  int      v4i;

constexpr int kB = 4096, kS = 256, kF = 32, kD = 64, kO = 16;
constexpr int kLabel  = kS / 2;            // 128
constexpr int kLLeft  = kLabel - 1;        // 127 left-scan steps (sites 1..127)
constexpr int kLRight = kS - 2 - kLabel;   // 126 right-scan steps (sites 129..254)
constexpr int kK = kF * kD;                // 2048 contraction length per step
constexpr int kTileB = 64;                 // batch rows per block
constexpr int kXRow  = 36;                 // LDS x-row stride in floats (TDM pad: 32+4)
constexpr size_t kStepElems = (size_t)kK * kD;  // 131072 weights per step

// ---------------------------------------------------------------------------
// One-time weight repack f32 -> f16, transposed to [step][col][k] so a WMMA
// B-fragment (lane holds 16 consecutive K for its column) is one aligned
// 32-byte load.
//   mode 0: WLT[l][e][d*32+f]  = W_left [l][d][f][e]
//   mode 1: WRT[j][d][e*32+f]  = W_right[j][d][f][e]
//   mode 2: WLAB[o][e][d*32+f] = W_label[d][f][e][o]
// ---------------------------------------------------------------------------
__global__ __launch_bounds__(256) void mps_convert_kernel(
    const float* __restrict__ in, hf* __restrict__ out, int mode, int total) {
  int idx = blockIdx.x * blockDim.x + threadIdx.x;
  if (idx >= total) return;
  int l   = idx >> 17;        // step index (or output channel o)
  int rem = idx & 131071;
  int c   = rem >> 11;        // column (output bond index)
  int k   = rem & 2047;
  int a   = k >> 5;           // non-f part of K index
  int f   = k & 31;
  long src;
  if (mode == 0)      src = (((long)(l * kD + a) * kF + f) * kD) + c; // d=a, e=c
  else if (mode == 1) src = (((long)(l * kD + c) * kF + f) * kD) + a; // d=c, e=a
  else                src = (((long)(a * kF + f) * kD + c) * kO) + l; // d=a, e=c, o=l
  out[idx] = (hf)in[src];
}

// ---------------------------------------------------------------------------
// Tensor Data Mover: DMA one x-site tile (64 batch rows x 32 floats, row
// stride S*F=8192 floats) into LDS, with TDM padding of 4 DWORDs per 32-DWORD
// row -> LDS row stride 36 floats (16B aligned, bank-conflict-free reads).
// D# group0: count=1, lds_addr, global_addr, type=2.
// D# group1: data_size=4B, pad_enable, pad_interval=32dw, pad_amount=4dw,
//            tensor_dim0=32, tensor_dim1=64, tile_dim0=32, tile_dim1=64,
//            tensor_dim0_stride=8192. Groups 2/3 unused (2-D tensor).
// clang-23 toolchain declares the 6-arg builtin (extra zero-filled group).
// ---------------------------------------------------------------------------
__device__ inline void tdm_load_x_tile(const float* gsrc, unsigned ldsoff) {
  unsigned long long ga = (unsigned long long)(uintptr_t)gsrc;
  v4u g0 = {1u, ldsoff, (unsigned)(ga & 0xffffffffu),
            (unsigned)((ga >> 32) & 0x01ffffffu) | (2u << 30)};
  //        w0: data_size=2<<16 | pad_en 1<<20 | pad_int 4<<22 | pad_amt 3<<25
  v8i g1  = {0x07120000, 32 << 16, 64 << 16, 32 << 16, 64, 8192, 0, 0};
  v4i gz4 = {0, 0, 0, 0};
  v8i gz8 = {0, 0, 0, 0, 0, 0, 0, 0};
  __builtin_amdgcn_tensor_load_to_lds(g0, g1, gz4, gz4, gz8, 0);
}

// ---------------------------------------------------------------------------
// Sweep kernel: blockIdx.y==0 -> left scan (produces v), ==1 -> right scan
// (produces r). 64 batch rows per block, 2 waves x 32 rows. Per step each
// wave runs a (32 x 2048) x (2048 x 64) GEMM via v_wmma_f32_16x16x32_f16.
// A K-chunk of 32 has a single bond index d, so the A operand is the
// x-fragment scaled by the per-row scalar v[row][d].
// x tiles are TDM-double-buffered across steps (s_wait_tensorcnt); weight
// B-fragments stream straight from L2 (compiler pipelines them).
// ---------------------------------------------------------------------------
__global__ __launch_bounds__(64) void mps_sweep_kernel(
    const float* __restrict__ x, const float* __restrict__ w0,
    const float* __restrict__ wlast, const hf* __restrict__ wlt,
    const hf* __restrict__ wrt, float* __restrict__ vout,
    float* __restrict__ rout) {
  __shared__ float vlds[kTileB][kD + 1];   // +1 pad kills d-broadcast conflicts
  __shared__ float xbuf[2][kTileB * kXRow];  // TDM double buffer (f32, padded rows)

  const int t    = threadIdx.x;      // 0..63
  const int lane = t & 31;
  const int wv   = t >> 5;           // wave id 0/1
  const int g    = lane >> 4;        // lane group (K sub-block of A/B frags)
  const int lm   = lane & 15;        // row (A) / column (B,C,D) within tile
  const int rowA = wv * 32;          // wave's base row inside the batch tile
  const int b0   = blockIdx.x * kTileB;
  const bool left = (blockIdx.y == 0);

  const v8f vzero = {0.f, 0.f, 0.f, 0.f, 0.f, 0.f, 0.f, 0.f};

  // ---- edge init: thread t owns batch row b0+t (f32 VALU, one-time tiny)
  {
    const int esite = left ? 0 : (kS - 1);
    const float* xr = x + ((size_t)(b0 + t) * kS + esite) * kF;
    float xv[kF];
#pragma unroll
    for (int f = 0; f < kF; ++f) xv[f] = xr[f];
#pragma unroll 4
    for (int d = 0; d < kD; ++d) {
      float s = 0.f;
#pragma unroll
      for (int f = 0; f < kF; ++f)
        s += xv[f] * (left ? w0[f * kD + d] : wlast[d * kF + f]);
      vlds[t][d] = s;
    }
  }

  const int nsteps = left ? kLLeft : kLRight;
  const hf* wbase  = left ? wlt : wrt;

  // prologue: DMA the first site's x tile into buffer 0
  if (wv == 0) {
    const int site0 = left ? 1 : (kS - 2);
    tdm_load_x_tile(x + ((size_t)b0 * kS + site0) * kF,
                    (unsigned)(uintptr_t)&xbuf[0][0]);
  }

  for (int it = 0; it < nsteps; ++it) {
    const int sw  = left ? it : (kLRight - 1 - it);  // weight index
    const int buf = it & 1;
    const hf* wst = wbase + (size_t)sw * kStepElems;

    // issue next tile into the other buffer (freed by B2 of step it-1),
    // then wait for the current tile; barrier publishes it to both waves
    if (wv == 0) {
      if (it + 1 < nsteps) {
        const int nsite = left ? (2 + it) : (kS - 3 - it);
        tdm_load_x_tile(x + ((size_t)b0 * kS + nsite) * kF,
                        (unsigned)(uintptr_t)&xbuf[buf ^ 1][0]);
        __builtin_amdgcn_s_wait_tensorcnt(1);
      } else {
        __builtin_amdgcn_s_wait_tensorcnt(0);
      }
    }
    __syncthreads();  // B1: xbuf[buf] valid for all waves

    // prefetch next step's weights (gfx1250 global_prefetch path)
    if (it + 1 < nsteps) {
      const hf* wn = wbase + (size_t)(left ? sw + 1 : sw - 1) * kStepElems;
      __builtin_prefetch(wn + t * 2048, 0, 0);
      __builtin_prefetch(wn + t * 2048 + 1024, 0, 0);
    }

    // assemble x A-fragments (f32 LDS -> f16) for the wave's two 16-row tiles
    // k order chosen here for A: group 0 holds K {0..15}, group 1 {16..31} (matches B)
    v16h xf[2];
#pragma unroll
    for (int rt = 0; rt < 2; ++rt) {
      const float* xr = &xbuf[buf][(rowA + rt * 16 + lm) * kXRow] + g * 16;
      v4f a0 = *(const v4f*)(xr + 0);
      v4f a1 = *(const v4f*)(xr + 4);
      v4f b0v = *(const v4f*)(xr + 8);
      v4f b1v = *(const v4f*)(xr + 12);
      v16h h;
      h[0] = (hf)a0.x;  h[1] = (hf)a0.y;  h[2] = (hf)a0.z;  h[3] = (hf)a0.w;
      h[4] = (hf)a1.x;  h[5] = (hf)a1.y;  h[6] = (hf)a1.z;  h[7] = (hf)a1.w;
      h[8] = (hf)b0v.x; h[9] = (hf)b0v.y; h[10] = (hf)b0v.z; h[11] = (hf)b0v.w;
      h[12] = (hf)b1v.x; h[13] = (hf)b1v.y; h[14] = (hf)b1v.z; h[15] = (hf)b1v.w;
      xf[rt] = h;
    }

    v8f acc[2][4];
#pragma unroll
    for (int rt = 0; rt < 2; ++rt)
#pragma unroll
      for (int ct = 0; ct < 4; ++ct) acc[rt][ct] = vzero;

#pragma unroll 4
    for (int kc = 0; kc < kD; ++kc) {  // K-chunk index == bond index d
      // B fragments: lane holds 16 consecutive K for its column -> aligned v16h
      v16h bf[4];
#pragma unroll
      for (int ct = 0; ct < 4; ++ct)
        bf[ct] = *(const v16h*)(wst + (size_t)(ct * 16 + lm) * kK + kc * 32 + g * 16);
#pragma unroll
      for (int rt = 0; rt < 2; ++rt) {
        hf h = (hf)vlds[rowA + rt * 16 + lm][kc];  // per-row scalar v[row][d]
        v16h a = xf[rt] * h;                       // 8x v_pk_mul_f16
#pragma unroll
        for (int ct = 0; ct < 4; ++ct)
          acc[rt][ct] = __builtin_amdgcn_wmma_f32_16x16x32_f16(
              false, a, false, bf[ct], (short)0, acc[rt][ct], false, false);
      }
    }

    // write back v_new (wave-private rows; reads all precede writes in-order)
#pragma unroll
    for (int rt = 0; rt < 2; ++rt)
#pragma unroll
      for (int ct = 0; ct < 4; ++ct)
#pragma unroll
        for (int i = 0; i < 8; ++i)
          vlds[rowA + rt * 16 + ((g == 0) ? i : (8 + i))][ct * 16 + lm] =
              acc[rt][ct][i];

    __syncthreads();  // B2: all waves done with xbuf[buf]; safe to reuse
  }

  // store carry state (f32) for the label kernel
  float* outp = (left ? vout : rout) + (size_t)(b0 + t) * kD;
#pragma unroll
  for (int d = 0; d < kD; d += 4) {
    v4f tmp = {vlds[t][d], vlds[t][d + 1], vlds[t][d + 2], vlds[t][d + 3]};
    *(v4f*)(outp + d) = tmp;
  }
}

// ---------------------------------------------------------------------------
// Label-site contraction: logits[b,o] = sum_e ( sum_{d,f} v_d x_f Wlab[d,f,e,o] ) r_e
// Per output channel o it is the same GEMM step; then an LDS dot with r.
// ---------------------------------------------------------------------------
__global__ __launch_bounds__(64) void mps_label_kernel(
    const float* __restrict__ x, const hf* __restrict__ wlab,
    const float* __restrict__ vin, const float* __restrict__ rin,
    float* __restrict__ out) {
  __shared__ float vlds[kTileB][kD + 1];
  __shared__ float rlds[kTileB][kD + 1];
  __shared__ float tlds[kTileB][kD + 1];
  __shared__ hf    xlds[kTileB][kF];

  const int t    = threadIdx.x;
  const int lane = t & 31;
  const int wv   = t >> 5;
  const int g    = lane >> 4;
  const int lm   = lane & 15;
  const int rowA = wv * 32;
  const int b0   = blockIdx.x * kTileB;

  const v8f vzero = {0.f, 0.f, 0.f, 0.f, 0.f, 0.f, 0.f, 0.f};

  // load carries (thread t -> row t, wave-private)
#pragma unroll 4
  for (int d = 0; d < kD; ++d) {
    vlds[t][d] = vin[(size_t)(b0 + t) * kD + d];
    rlds[t][d] = rin[(size_t)(b0 + t) * kD + d];
  }
  // stage label-site x rows
  {
    const float* xr = x + ((size_t)(b0 + t) * kS + kLabel) * kF;
#pragma unroll
    for (int f = 0; f < kF; f += 4) {
      v4f xv = *(const v4f*)(xr + f);
      xlds[t][f + 0] = (hf)xv.x;
      xlds[t][f + 1] = (hf)xv.y;
      xlds[t][f + 2] = (hf)xv.z;
      xlds[t][f + 3] = (hf)xv.w;
    }
  }

  v16h xf[2];
#pragma unroll
  for (int rt = 0; rt < 2; ++rt) {
    const hf* xp = &xlds[rowA + rt * 16 + lm][0] + g * 16;
    v8h lo = *(const v8h*)xp;
    v8h hi = *(const v8h*)(xp + 8);
    xf[rt] = __builtin_shufflevector(lo, hi, 0, 1, 2, 3, 4, 5, 6, 7,
                                     8, 9, 10, 11, 12, 13, 14, 15);
  }

  for (int o = 0; o < kO; ++o) {
    const hf* wst = wlab + (size_t)o * kStepElems;

    v8f acc[2][4];
#pragma unroll
    for (int rt = 0; rt < 2; ++rt)
#pragma unroll
      for (int ct = 0; ct < 4; ++ct) acc[rt][ct] = vzero;

#pragma unroll 4
    for (int kc = 0; kc < kD; ++kc) {
      v16h bf[4];
#pragma unroll
      for (int ct = 0; ct < 4; ++ct)
        bf[ct] = *(const v16h*)(wst + (size_t)(ct * 16 + lm) * kK + kc * 32 + g * 16);
#pragma unroll
      for (int rt = 0; rt < 2; ++rt) {
        hf h = (hf)vlds[rowA + rt * 16 + lm][kc];
        v16h a = xf[rt] * h;
#pragma unroll
        for (int ct = 0; ct < 4; ++ct)
          acc[rt][ct] = __builtin_amdgcn_wmma_f32_16x16x32_f16(
              false, a, false, bf[ct], (short)0, acc[rt][ct], false, false);
      }
    }

    // scatter t = A@W_o to LDS, then dot with r per row (all intra-wave)
#pragma unroll
    for (int rt = 0; rt < 2; ++rt)
#pragma unroll
      for (int ct = 0; ct < 4; ++ct)
#pragma unroll
        for (int i = 0; i < 8; ++i)
          tlds[rowA + rt * 16 + ((g == 0) ? i : (8 + i))][ct * 16 + lm] =
              acc[rt][ct][i];

    float s = 0.f;
#pragma unroll 8
    for (int e = 0; e < kD; ++e) s += tlds[t][e] * rlds[t][e];
    out[(size_t)(b0 + t) * kO + o] = s;
  }
}

// ---------------------------------------------------------------------------
// Workspace layout (bytes):
//   WLT16 : 127*131072*2 = 33,292,288
//   WRT16 : 126*131072*2 = 33,030,144
//   WLAB16:  16*131072*2 =  4,194,304
//   vout  : 4096*64*4    =  1,048,576
//   rout  : 4096*64*4    =  1,048,576     total ~72.6 MB
// ---------------------------------------------------------------------------
extern "C" void kernel_launch(void* const* d_in, const int* in_sizes, int n_in,
                              void* d_out, int out_size, void* d_ws,
                              size_t ws_size, hipStream_t stream) {
  const float* x     = (const float*)d_in[0];
  const float* w0    = (const float*)d_in[1];
  const float* wleft = (const float*)d_in[2];
  const float* wlab  = (const float*)d_in[3];
  const float* wrigh = (const float*)d_in[4];
  const float* wlast = (const float*)d_in[5];

  char* ws = (char*)d_ws;
  hf*    WLT  = (hf*)(ws + 0);
  hf*    WRT  = (hf*)(ws + 33292288);
  hf*    WLAB = (hf*)(ws + 66322432);
  float* vout = (float*)(ws + 70516736);
  float* rout = (float*)(ws + 71565312);

  const int tot0 = kLLeft * (int)kStepElems;   // 16,646,144
  const int tot1 = kLRight * (int)kStepElems;  // 16,515,072
  const int tot2 = kO * (int)kStepElems;       //  2,097,152

  mps_convert_kernel<<<(tot0 + 255) / 256, 256, 0, stream>>>(wleft, WLT, 0, tot0);
  mps_convert_kernel<<<(tot1 + 255) / 256, 256, 0, stream>>>(wrigh, WRT, 1, tot1);
  mps_convert_kernel<<<(tot2 + 255) / 256, 256, 0, stream>>>(wlab, WLAB, 2, tot2);

  mps_sweep_kernel<<<dim3(kB / kTileB, 2), 64, 0, stream>>>(
      x, w0, wlast, WLT, WRT, vout, rout);

  mps_label_kernel<<<kB / kTileB, 64, 0, stream>>>(x, WLAB, vout, rout,
                                                   (float*)d_out);
}
